// SetConvDecoder_10806137717278
// MI455X (gfx1250) — matrix operand order repaired and checked by reading, not compiled, not run on hardware
//
#include <hip/hip_runtime.h>

// Fused SetConv decoder for MI455X (gfx1250, wave32).
// out[b,t,d,k] = sum_g exp(-0.5*||(xt[b,t]-xg[b,g])/ls[:,k]||^2) * zg[b,g,d]
// 8 fused GEMMs (M=4096,K=4096,N=128); RBF weights generated on the fly in
// WMMA A-fragment layout; v_wmma_f32_16x16x32_f16 mainloop; double-buffered
// LDS staging of the z operand; software-pipelined B-fragment loads.

typedef __attribute__((ext_vector_type(16))) _Float16 v16h;
typedef __attribute__((ext_vector_type(8)))  float    v8f;

#define B_   4
#define NT_  4096
#define NG_  4096
#define DZ_  128
#define NK_  2
#define DX_  2
#define KC   64     // K (grid) chunk staged in LDS
#define MBLK 256    // target rows per workgroup (8 waves x 32 rows)
#define NCHUNK (NG_ / KC)

__global__ __launch_bounds__(256)
void setconv_decoder_wmma(const float* __restrict__ x_grid,
                          const float* __restrict__ z_grid,
                          const float* __restrict__ xt,
                          const float* __restrict__ lsp,
                          float* __restrict__ out)
{
    // Double-buffered staging. zsh: z chunk as f16, transposed [d][k] so a
    // B fragment is 16 contiguous halfs per lane (two ds_load_b128).
    __shared__ __attribute__((aligned(32))) _Float16 zsh[2][DZ_ * KC]; // 32 KB
    __shared__ float4 gsh[2][KC];                                     //  2 KB

    const int tid  = threadIdx.x;
    const int lane = tid & 31;
    const int wv   = tid >> 5;

    const int mt = blockIdx.x;          // 0..15 : M tile
    const int b  = blockIdx.y / NK_;    // batch
    const int k  = blockIdx.y % NK_;    // kernel index

    const float LOG2E = 1.44269504088896340736f;

    // lengthscale = 1e-5 + softplus(param), per (dx, k)
    const float p0  = lsp[0 * NK_ + k];
    const float p1  = lsp[1 * NK_ + k];
    const float ls0 = 1e-5f + ((p0 > 20.0f) ? p0 : log1pf(__expf(p0)));
    const float ls1 = 1e-5f + ((p1 > 20.0f) ? p1 : log1pf(__expf(p1)));
    const float inv0 = 1.0f / ls0;
    const float inv1 = 1.0f / ls1;

    // Two target rows per lane (A fragments m and m+16 of this wave's strip).
    // -0.5*d2 = a.b - 0.5|a|^2 - 0.5|b|^2 ; fold log2(e) so we can use exp2.
    const int   r0   = mt * MBLK + wv * 32 + (lane & 15);
    const float x00  = xt[(b * NT_ + r0) * DX_ + 0] * inv0;
    const float x01  = xt[(b * NT_ + r0) * DX_ + 1] * inv1;
    const float x10  = xt[(b * NT_ + r0 + 16) * DX_ + 0] * inv0;
    const float x11  = xt[(b * NT_ + r0 + 16) * DX_ + 1] * inv1;
    const float a0k0 = LOG2E * x00, a1k0 = LOG2E * x01;
    const float a0k1 = LOG2E * x10, a1k1 = LOG2E * x11;
    const float ca0  = -0.5f * LOG2E * (x00 * x00 + x01 * x01);
    const float ca1  = -0.5f * LOG2E * (x10 * x10 + x11 * x11);

    v8f acc0[8], acc1[8];
#pragma unroll
    for (int n = 0; n < 8; ++n)
#pragma unroll
        for (int j = 0; j < 8; ++j) { acc0[n][j] = 0.0f; acc1[n][j] = 0.0f; }

    const float* zb = z_grid + b * NG_ * DZ_;
    const float* xb = x_grid + b * NG_ * DX_;

    // z-chunk staging assignment: 128 threads per half-chunk of K.
    const int dcol  = tid & 127;              // column (d) this thread stages
    const int kbase = (tid >> 7) * (KC / 2);  // 0 or 32

    auto stage = [&](int g0, int buf) {
#pragma unroll
        for (int j = 0; j < KC / 2; j += 2) {
            float f0 = zb[(g0 + kbase + j)     * DZ_ + dcol];
            float f1 = zb[(g0 + kbase + j + 1) * DZ_ + dcol];
            zsh[buf][dcol * KC + kbase + j]     = (_Float16)f0;
            zsh[buf][dcol * KC + kbase + j + 1] = (_Float16)f1;
        }
        if (tid < KC) {
            int   g  = g0 + tid;
            float b0 = xb[g * DX_ + 0] * inv0;
            float b1 = xb[g * DX_ + 1] * inv1;
            float cg = -0.5f * LOG2E * (b0 * b0 + b1 * b1);
            gsh[buf][tid] = make_float4(b0, b1, cg, 0.0f);
        }
    };

    stage(0, 0);
    __syncthreads();

    for (int it = 0; it < NCHUNK; ++it) {
        const int cur = it & 1;
        if (it + 1 < NCHUNK) stage((it + 1) * KC, cur ^ 1);  // prefetch next

        const float4*   gc = gsh[cur];
        const _Float16* zc = zsh[cur];

#pragma unroll
        for (int ks = 0; ks < KC / 32; ++ks) {
            const int kofs = ks * 32;

            // Two A fragments (16x32 f16), rows r0 and r0+16; lane l holds row
            // l&15; element e -> K = (lane>>4)*16 + e. The gsh
            // read is shared between both rows.
            const int klo = (lane >> 4) * 16;
            v16h af0{}, af1{};
#pragma unroll
            for (int e = 0; e < 16; ++e) {
                const int    kk = kofs + klo + e;
                const float4 gv = gc[kk];
                const float arg0 = fmaf(a0k0, gv.x, fmaf(a1k0, gv.y, ca0 + gv.z));
                const float arg1 = fmaf(a0k1, gv.x, fmaf(a1k1, gv.y, ca1 + gv.z));
                af0[e] = (_Float16)__builtin_amdgcn_exp2f(arg0);
                af1[e] = (_Float16)__builtin_amdgcn_exp2f(arg1);
            }

            // 8 N-tiles; each B fragment feeds two WMMAs (rows m, m+16).
            // Software-pipeline the B loads one tile ahead so the ds_load pair
            // for tile n+1 is in flight while tile n's WMMAs execute.
            const int kh    = kofs + (lane >> 4) * 16;
            const int cbase = lane & 15;
            v16h bf = *(const v16h*)&zc[cbase * KC + kh];
#pragma unroll
            for (int n = 0; n < 8; ++n) {
                v16h bf_next = bf;
                if (n < 7)
                    bf_next = *(const v16h*)&zc[((n + 1) * 16 + cbase) * KC + kh];
                acc0[n] = __builtin_amdgcn_wmma_f32_16x16x32_f16(
                    false, af0, false, bf, (short)0, acc0[n], false, false);
                acc1[n] = __builtin_amdgcn_wmma_f32_16x16x32_f16(
                    false, af1, false, bf, (short)0, acc1[n], false, false);
                bf = bf_next;
            }
        }
        __syncthreads();
    }

    // C layout: VGPR j, lanes 0-15 -> M=j, N=lane; lanes 16-31 -> M=8+j, N=lane-16.
    float* ob = out + b * NT_ * DZ_ * NK_;
    const int rbase = mt * MBLK + wv * 32 + (lane >> 4) * 8;
#pragma unroll
    for (int n = 0; n < 8; ++n) {
        const int col = n * 16 + (lane & 15);
#pragma unroll
        for (int j = 0; j < 8; ++j) {
            ob[((rbase + j) * DZ_ + col) * NK_ + k]      = acc0[n][j];
            ob[((rbase + 16 + j) * DZ_ + col) * NK_ + k] = acc1[n][j];
        }
    }
}

extern "C" void kernel_launch(void* const* d_in, const int* in_sizes, int n_in,
                              void* d_out, int out_size, void* d_ws, size_t ws_size,
                              hipStream_t stream) {
    const float* x_grid = (const float*)d_in[0];
    const float* z_grid = (const float*)d_in[1];
    const float* xt     = (const float*)d_in[2];
    const float* lsp    = (const float*)d_in[3];
    float* out = (float*)d_out;

    dim3 grid(NT_ / MBLK, B_ * NK_);   // 16 x 8 = 128 workgroups
    dim3 block(256);                   // 8 waves (wave32)
    setconv_decoder_wmma<<<grid, block, 0, stream>>>(x_grid, z_grid, xt, lsp, out);
}
